// CombinedModelGATMLP_33225867002446
// MI455X (gfx1250) — compile-verified, matrix operand order checked by reading, not run on hardware
//
#include <hip/hip_runtime.h>
#include <hip/hip_bf16.h>

// ---------------- types ----------------
typedef __attribute__((ext_vector_type(16))) __bf16 v16bf;
typedef __attribute__((ext_vector_type(8)))  float  v8f;

#define N_NODES 50000
#define N_EDGES 800000
#define N_GRAPH 1000
#define D_IN    64
#define HID     64
#define GOUT    256
#define FP_DIM  2048
#define H1_DIM  1024
#define H2_DIM  512
#define MOUT    256
#define D1_DIM  512

// ---------------- ordered-float helpers (for atomic segment max) ----------------
__device__ __forceinline__ unsigned f2o(float f) {
  unsigned u = __float_as_uint(f);
  return (u & 0x80000000u) ? ~u : (u | 0x80000000u);
}
__device__ __forceinline__ float o2f(unsigned u) {
  u = (u & 0x80000000u) ? (u & 0x7FFFFFFFu) : ~u;
  return __uint_as_float(u);
}

// ---------------- fills ----------------
__global__ void fill_f32(float* p, float v, long n) {
  long i = (long)blockIdx.x * blockDim.x + threadIdx.x;
  if (i < n) p[i] = v;
}
__global__ void fill_u32(unsigned* p, unsigned v, long n) {
  long i = (long)blockIdx.x * blockDim.x + threadIdx.x;
  if (i < n) p[i] = v;
}

// ---------------- WMMA bf16 fragment load (unconditional, pointer pre-clamped) ----------------
// 16-bit A/B layout (wave32): lane = half*16 + idx, row = idx.
// VGPR v (elements 2v,2v+1): v=0..3 -> K = half*8 + [0..7]; v=4..7 -> K = 16 + half*8 + [0..7]
__device__ __forceinline__ v16bf load_frag_bf16(const float* __restrict__ rp,
                                                int k0, int half) {
  const float4* pa = reinterpret_cast<const float4*>(rp + k0 + half * 8);
  const float4* pb = reinterpret_cast<const float4*>(rp + k0 + 16 + half * 8);
  float4 a0 = pa[0], a1 = pa[1];
  float4 b0 = pb[0], b1 = pb[1];
  v16bf f;
  f[0]  = (__bf16)a0.x; f[1]  = (__bf16)a0.y; f[2]  = (__bf16)a0.z; f[3]  = (__bf16)a0.w;
  f[4]  = (__bf16)a1.x; f[5]  = (__bf16)a1.y; f[6]  = (__bf16)a1.z; f[7]  = (__bf16)a1.w;
  f[8]  = (__bf16)b0.x; f[9]  = (__bf16)b0.y; f[10] = (__bf16)b0.z; f[11] = (__bf16)b0.w;
  f[12] = (__bf16)b1.x; f[13] = (__bf16)b1.y; f[14] = (__bf16)b1.z; f[15] = (__bf16)b1.w;
  return f;
}

__device__ __forceinline__ v8f wmma_bf16(v16bf a, v16bf b, v8f c) {
  return __builtin_amdgcn_wmma_f32_16x16x32_bf16(false, a, false, b, (short)0, c,
                                                 false, false);
}

// ---------------- generic GEMM: C[M,N] = A[M,K] @ B[N,K]^T (+bias, opt ReLU) ----------------
// One wave per 32x32 macro-tile of C (2x2 WMMA tiles, A/B fragment reuse).
// 4 waves per block. K must be a multiple of 32; M and N tails handled by
// clamped loads (garbage rows/cols are never stored).
__global__ void gemm_wmma_bf16(const float* __restrict__ A, const float* __restrict__ B,
                               const float* __restrict__ bias, float* __restrict__ C,
                               int M, int N, int K, int ldc, int relu) {
  int wave = threadIdx.x >> 5;
  int lane = threadIdx.x & 31;
  int tm = (blockIdx.y * 4 + wave) * 32;    // M base of macro tile
  int tn = blockIdx.x * 32;                 // N base of macro tile
  if (tm >= M) return;                      // wave-uniform: EXEC all-ones for WMMA

  int half = lane >> 4;
  int idx  = lane & 15;

  // clamped row pointers (OOB lanes read a valid row; results land in unstored C)
  const float* a0 = A + (size_t)min(tm + idx,      M - 1) * K;
  const float* a1 = A + (size_t)min(tm + 16 + idx, M - 1) * K;
  const float* b0 = B + (size_t)min(tn + idx,      N - 1) * K;
  const float* b1 = B + (size_t)min(tn + 16 + idx, N - 1) * K;

  v8f acc00 = {}, acc01 = {}, acc10 = {}, acc11 = {};
  for (int k0 = 0; k0 < K; k0 += 32) {
    v16bf fa0 = load_frag_bf16(a0, k0, half);
    v16bf fa1 = load_frag_bf16(a1, k0, half);
    v16bf fb0 = load_frag_bf16(b0, k0, half);
    v16bf fb1 = load_frag_bf16(b1, k0, half);
    acc00 = wmma_bf16(fa0, fb0, acc00);
    acc01 = wmma_bf16(fa0, fb1, acc01);
    acc10 = wmma_bf16(fa1, fb0, acc10);
    acc11 = wmma_bf16(fa1, fb1, acc11);
  }

  // C/D layout per 16x16 tile: col = idx, row (acc element r) = r + half*8
  v8f accs[4] = {acc00, acc01, acc10, acc11};
#pragma unroll
  for (int mi = 0; mi < 2; ++mi) {
#pragma unroll
    for (int ni = 0; ni < 2; ++ni) {
      v8f acc = accs[mi * 2 + ni];
      int col = tn + ni * 16 + idx;
      if (col < N) {
        float bv = bias ? bias[col] : 0.0f;
#pragma unroll
        for (int r = 0; r < 8; ++r) {
          int row = tm + mi * 16 + r + half * 8;
          if (row < M) {
            float v = acc[r] + bv;
            if (relu) v = fmaxf(v, 0.0f);
            C[(size_t)row * ldc + col] = v;
          }
        }
      }
    }
  }
}

// ---------------- GAT pieces ----------------
__global__ void attn_dots(const float* __restrict__ xl, const float* __restrict__ att_src,
                          const float* __restrict__ att_dst, float* as_, float* ad_, int Nn) {
  int n = blockIdx.x * blockDim.x + threadIdx.x;
  if (n >= Nn) return;
  const float* r = xl + (size_t)n * HID;
  float s = 0.0f, d = 0.0f;
#pragma unroll
  for (int k = 0; k < HID; ++k) { s += r[k] * att_src[k]; d += r[k] * att_dst[k]; }
  as_[n] = s; ad_[n] = d;
}

__global__ void edge_scores(const int* __restrict__ ei, const float* __restrict__ as_,
                            const float* __restrict__ ad_, float* __restrict__ e,
                            unsigned* __restrict__ emax, int E, int Nn) {
  int i = blockIdx.x * blockDim.x + threadIdx.x;
  if (i >= E + Nn) return;
  int s, d;
  if (i < E) { s = ei[i]; d = ei[E + i]; } else { s = d = i - E; }
  float v = as_[s] + ad_[d];
  v = v > 0.0f ? v : 0.2f * v;              // leaky relu
  e[i] = v;
  atomicMax(&emax[d], f2o(v));
}

__global__ void edge_exp(const int* __restrict__ ei, float* __restrict__ e,
                         const unsigned* __restrict__ emax, float* __restrict__ denom,
                         int E, int Nn) {
  int i = blockIdx.x * blockDim.x + threadIdx.x;
  if (i >= E + Nn) return;
  int d = (i < E) ? ei[E + i] : (i - E);
  float ex = __expf(e[i] - o2f(emax[d]));
  e[i] = ex;
  atomicAdd(&denom[d], ex);
}

// grid = E+N blocks, 64 threads (one per channel)
__global__ void edge_agg(const int* __restrict__ ei, const float* __restrict__ e,
                         const float* __restrict__ denom, const float* __restrict__ xl,
                         float* __restrict__ outAgg, int E) {
  int i = blockIdx.x;
  int c = threadIdx.x;
  int s, d;
  if (i < E) { s = ei[i]; d = ei[E + i]; } else { s = d = i - E; }
  float alpha = e[i] / denom[d];
  atomicAdd(&outAgg[(size_t)d * HID + c], xl[(size_t)s * HID + c] * alpha);
}

__global__ void relu_pool(const float* __restrict__ outAgg, const float* __restrict__ gat_bias,
                          const int* __restrict__ batch, float* __restrict__ sums,
                          float* __restrict__ cnt, int Nn) {
  int i = blockIdx.x * blockDim.x + threadIdx.x;
  if (i >= Nn * HID) return;
  int n = i >> 6, c = i & 63;
  float v = fmaxf(outAgg[i] + gat_bias[c], 0.0f);
  int b = batch[n];
  atomicAdd(&sums[(size_t)b * HID + c], v);
  if (c == 0) atomicAdd(&cnt[b], 1.0f);
}

__global__ void pool_div(const float* __restrict__ sums, const float* __restrict__ cnt,
                         float* __restrict__ pooled, int Gn) {
  int i = blockIdx.x * blockDim.x + threadIdx.x;
  if (i >= Gn * HID) return;
  pooled[i] = sums[i] / fmaxf(cnt[i >> 6], 1.0f);
}

// ---------------- BatchNorm (training mode, biased var) ----------------
__global__ void bn_stats(const float* __restrict__ X, int Gn, int C,
                         float* __restrict__ mu, float* __restrict__ rsig) {
  int c = blockIdx.x;
  __shared__ float s1[256], s2[256];
  float s = 0.0f, sq = 0.0f;
  for (int g = threadIdx.x; g < Gn; g += blockDim.x) {
    float v = X[(size_t)g * C + c];
    s += v; sq += v * v;
  }
  s1[threadIdx.x] = s; s2[threadIdx.x] = sq;
  __syncthreads();
  for (int off = blockDim.x >> 1; off; off >>= 1) {
    if ((int)threadIdx.x < off) {
      s1[threadIdx.x] += s1[threadIdx.x + off];
      s2[threadIdx.x] += s2[threadIdx.x + off];
    }
    __syncthreads();
  }
  if (threadIdx.x == 0) {
    float m = s1[0] / (float)Gn;
    float var = s2[0] / (float)Gn - m * m;
    mu[c] = m;
    rsig[c] = rsqrtf(var + 1e-5f);
  }
}

__global__ void bn_apply_relu(float* __restrict__ X, int Gn, int C,
                              const float* __restrict__ mu, const float* __restrict__ rsig,
                              const float* __restrict__ gamma, const float* __restrict__ beta) {
  int i = blockIdx.x * blockDim.x + threadIdx.x;
  if (i >= Gn * C) return;
  int c = i % C;
  float v = (X[i] - mu[c]) * rsig[c] * gamma[c] + beta[c];
  X[i] = fmaxf(v, 0.0f);
}

// ---------------- final head: out[g] = bn(z[g,:]) . Wl + bl ----------------
__global__ void final_head(const float* __restrict__ z, const float* __restrict__ mu,
                           const float* __restrict__ rsig, const float* __restrict__ gf,
                           const float* __restrict__ bef, const float* __restrict__ Wl,
                           const float* __restrict__ bl, float* __restrict__ out) {
  int g = blockIdx.x;
  __shared__ float red[128];
  float s = 0.0f;
  for (int j = threadIdx.x; j < D1_DIM; j += blockDim.x) {
    float v = (z[(size_t)g * D1_DIM + j] - mu[j]) * rsig[j] * gf[j] + bef[j];
    s += v * Wl[j];
  }
  red[threadIdx.x] = s;
  __syncthreads();
  for (int off = blockDim.x >> 1; off; off >>= 1) {
    if ((int)threadIdx.x < off) red[threadIdx.x] += red[threadIdx.x + off];
    __syncthreads();
  }
  if (threadIdx.x == 0) out[g] = red[0] + bl[0];
}

// ---------------- driver ----------------
extern "C" void kernel_launch(void* const* d_in, const int* in_sizes, int n_in,
                              void* d_out, int out_size, void* d_ws, size_t ws_size,
                              hipStream_t stream) {
  const float* x        = (const float*)d_in[0];
  const int*   ei       = (const int*)  d_in[1];
  const int*   batch    = (const int*)  d_in[2];
  const float* fps      = (const float*)d_in[3];
  const float* Wg       = (const float*)d_in[4];
  const float* att_src  = (const float*)d_in[5];
  const float* att_dst  = (const float*)d_in[6];
  const float* gat_bias = (const float*)d_in[7];
  const float* Wlg      = (const float*)d_in[8];
  const float* blg      = (const float*)d_in[9];
  const float* W1       = (const float*)d_in[10];
  const float* b1       = (const float*)d_in[11];
  const float* g1       = (const float*)d_in[12];
  const float* be1      = (const float*)d_in[13];
  const float* W2       = (const float*)d_in[14];
  const float* b2       = (const float*)d_in[15];
  const float* g2       = (const float*)d_in[16];
  const float* be2      = (const float*)d_in[17];
  const float* W3       = (const float*)d_in[18];
  const float* b3       = (const float*)d_in[19];
  const float* Wf       = (const float*)d_in[20];
  const float* bfv      = (const float*)d_in[21];
  const float* gf       = (const float*)d_in[22];
  const float* bef      = (const float*)d_in[23];
  const float* Wl       = (const float*)d_in[24];
  const float* bl       = (const float*)d_in[25];
  float* out = (float*)d_out;

  // carve workspace
  uintptr_t p = (uintptr_t)d_ws;
  auto carve = [&](size_t nfloats) -> float* {
    p = (p + 255) & ~(uintptr_t)255;
    float* r = (float*)p;
    p += nfloats * sizeof(float);
    return r;
  };
  float*    xl     = carve((size_t)N_NODES * HID);
  float*    as_    = carve(N_NODES);
  float*    ad_    = carve(N_NODES);
  float*    ebuf   = carve(N_EDGES + N_NODES);
  unsigned* emax   = (unsigned*)carve(N_NODES);
  float*    denom  = carve(N_NODES);
  float*    outAgg = carve((size_t)N_NODES * HID);
  float*    sums   = carve((size_t)N_GRAPH * HID);
  float*    cnt    = carve(N_GRAPH);
  float*    pooled = carve((size_t)N_GRAPH * HID);
  float*    cbuf   = carve((size_t)N_GRAPH * (GOUT + MOUT));   // [G, 512] concat
  float*    h1     = carve((size_t)N_GRAPH * H1_DIM);
  float*    h2     = carve((size_t)N_GRAPH * H2_DIM);
  float*    zbuf   = carve((size_t)N_GRAPH * D1_DIM);
  float*    mu     = carve(H1_DIM);
  float*    rsig   = carve(H1_DIM);

  auto gemm = [&](const float* A, const float* B, const float* bias, float* C,
                  int M, int N, int K, int ldc, int relu) {
    dim3 grid((N + 31) / 32, ((M + 31) / 32 + 3) / 4);
    gemm_wmma_bf16<<<grid, 128, 0, stream>>>(A, B, bias, C, M, N, K, ldc, relu);
  };

  const int TEN = N_EDGES + N_NODES;

  // ---- init scratch that is accumulated into ----
  fill_u32<<<(N_NODES + 255) / 256, 256, 0, stream>>>(emax, 0x007FFFFFu /*f2o(-inf)*/, N_NODES);
  fill_f32<<<(N_NODES + 255) / 256, 256, 0, stream>>>(denom, 0.0f, N_NODES);
  fill_f32<<<((long)N_NODES * HID + 255) / 256, 256, 0, stream>>>(outAgg, 0.0f, (long)N_NODES * HID);
  fill_f32<<<((long)N_GRAPH * HID + 255) / 256, 256, 0, stream>>>(sums, 0.0f, (long)N_GRAPH * HID);
  fill_f32<<<(N_GRAPH + 255) / 256, 256, 0, stream>>>(cnt, 0.0f, N_GRAPH);

  // ---- GATConv ----
  gemm(x, Wg, nullptr, xl, N_NODES, HID, D_IN, HID, 0);                 // xl = x @ Wg^T
  attn_dots<<<(N_NODES + 255) / 256, 256, 0, stream>>>(xl, att_src, att_dst, as_, ad_, N_NODES);
  edge_scores<<<(TEN + 255) / 256, 256, 0, stream>>>(ei, as_, ad_, ebuf, emax, N_EDGES, N_NODES);
  edge_exp<<<(TEN + 255) / 256, 256, 0, stream>>>(ei, ebuf, emax, denom, N_EDGES, N_NODES);
  edge_agg<<<TEN, 64, 0, stream>>>(ei, ebuf, denom, xl, outAgg, N_EDGES);

  // ---- ReLU + global_mean_pool + lin ----
  relu_pool<<<((long)N_NODES * HID + 255) / 256, 256, 0, stream>>>(outAgg, gat_bias, batch, sums, cnt, N_NODES);
  pool_div<<<((long)N_GRAPH * HID + 255) / 256, 256, 0, stream>>>(sums, cnt, pooled, N_GRAPH);
  gemm(pooled, Wlg, blg, cbuf, N_GRAPH, GOUT, HID, GOUT + MOUT, 0);     // gnn_out -> cbuf[:, :256]

  // ---- MLP on fingerprints ----
  gemm(fps, W1, b1, h1, N_GRAPH, H1_DIM, FP_DIM, H1_DIM, 0);
  bn_stats<<<H1_DIM, 256, 0, stream>>>(h1, N_GRAPH, H1_DIM, mu, rsig);
  bn_apply_relu<<<((long)N_GRAPH * H1_DIM + 255) / 256, 256, 0, stream>>>(h1, N_GRAPH, H1_DIM, mu, rsig, g1, be1);
  gemm(h1, W2, b2, h2, N_GRAPH, H2_DIM, H1_DIM, H2_DIM, 0);
  bn_stats<<<H2_DIM, 256, 0, stream>>>(h2, N_GRAPH, H2_DIM, mu, rsig);
  bn_apply_relu<<<((long)N_GRAPH * H2_DIM + 255) / 256, 256, 0, stream>>>(h2, N_GRAPH, H2_DIM, mu, rsig, g2, be2);
  gemm(h2, W3, b3, cbuf + GOUT, N_GRAPH, MOUT, H2_DIM, GOUT + MOUT, 0); // mlp_out -> cbuf[:, 256:]

  // ---- fusion head ----
  gemm(cbuf, Wf, bfv, zbuf, N_GRAPH, D1_DIM, GOUT + MOUT, D1_DIM, 1);   // relu(c @ Wf^T + bf)
  bn_stats<<<D1_DIM, 256, 0, stream>>>(zbuf, N_GRAPH, D1_DIM, mu, rsig);
  final_head<<<N_GRAPH, 128, 0, stream>>>(zbuf, mu, rsig, gf, bef, Wl, bl, out);
}
